// HungarianPITLoss_53747220742874
// MI455X (gfx1250) — matrix operand order repaired and checked by reading, not compiled, not run on hardware
//
#include <hip/hip_runtime.h>
#include <hip/hip_bf16.h>

// HungarianPITLoss for B=64, T=8000, S=10.
// pairwise[b,i,j] = -( M[b,i,j] + S1[b,i] ) / T  with
//   M[b,i,j]  = sum_t logit(pred[b,t,i]) * tgt[b,t,j]   (16x16xK WMMA GEMM, S padded to 16)
//   S1[b,i]   = sum_t log(1 - pred[b,t,i])
// then a 10x10 Jonker-Volgenant assignment per batch and a mean over the 640 matches.
//
// Data path: coalesced global_load_b128 -> per-wave LDS tile -> ds_load gather into
// WMMA register layout -> v_wmma_f32_16x16x32_f16 accumulate (K=8000 over 250 tiles).

typedef __attribute__((ext_vector_type(16))) _Float16 v16h;
typedef __attribute__((ext_vector_type(8)))  float    v8f;

#define BB 64
#define TT 8000
#define SS 10
#define NWAVES 16                       // 512 threads = 16 wave32 waves per block
#define KTILES (TT / 32)                // 250 K-tiles of 32 timesteps
#define NITER  ((KTILES + NWAVES - 1) / NWAVES)   // 16 uniform iterations
#define TILE_F4 80                      // 32 timesteps * 10 floats = 320 f32 = 80 float4

__global__ __launch_bounds__(32 * NWAVES)
void pit_pairwise_wmma_kernel(const float* __restrict__ pred,
                              const float* __restrict__ tgt,
                              float* __restrict__ Mws,    // [B][16][16]
                              float* __restrict__ S1ws) { // [B][16]
    __shared__ float4 sA4[NWAVES * TILE_F4];   // 20 KB: pred tiles, one slab per wave
    __shared__ float4 sB4[NWAVES * TILE_F4];   // 20 KB: target tiles
    __shared__ float  sAcc[NWAVES * 256];      // 16 KB: per-wave WMMA accumulators
    __shared__ float  sS1[NWAVES * 32];        //  2 KB: per-lane log1p partials

    const int b    = blockIdx.x;
    const int wave = threadIdx.x >> 5;
    const int lane = threadIdx.x & 31;
    const int m    = lane & 15;        // A row (i) / B col (j), padded 10->16
    const int h    = lane >> 4;        // lane half-group
    const int mc   = (m < SS) ? m : 0; // clamped column (branchless padding)
    const bool valid = (m < SS);

    const float* pb = pred + (size_t)b * TT * SS;
    const float* tb = tgt  + (size_t)b * TT * SS;
    const float* As = reinterpret_cast<const float*>(&sA4[wave * TILE_F4]);
    const float* Bs = reinterpret_cast<const float*>(&sB4[wave * TILE_F4]);

    v8f   acc = {};
    float s1  = 0.0f;

    for (int it = 0; it < NITER; ++it) {
        const int tile   = it * NWAVES + wave;
        const bool active = (tile < KTILES);

        if (active) {
            // --- Stage this wave's 1280B tile slabs with wide coalesced loads ---
            const float4* pv4 = reinterpret_cast<const float4*>(pb) + (size_t)tile * TILE_F4;
            const float4* tv4 = reinterpret_cast<const float4*>(tb) + (size_t)tile * TILE_F4;
#pragma unroll
            for (int pass = 0; pass < 3; ++pass) {
                const int idx = pass * 32 + lane;
                if (idx < TILE_F4) {
                    sA4[wave * TILE_F4 + idx] = pv4[idx];
                    sB4[wave * TILE_F4 + idx] = tv4[idx];
                }
            }
            // Prefetch this wave's next tile (20 lanes x 64B covers the 1280B slab).
            if (tile + NWAVES < KTILES) {
                __builtin_prefetch((const char*)(pv4 + NWAVES * TILE_F4) + lane * 64, 0, 1);
                __builtin_prefetch((const char*)(tv4 + NWAVES * TILE_F4) + lane * 64, 0, 1);
            }
        }
        __syncthreads();

        if (active) {
            // --- Gather into WMMA register layout (branchless) + accumulate ---
            v16h a;   // A: 16x32 f16, A[m][k] = logit(pred[t0+k][m])
            v16h bt;  // B: 32x16 f16, B[k][n] = tgt[t0+k][n]
#pragma unroll
            for (int p = 0; p < 8; ++p) {
                // A, contiguous k order (k=16h+s): VGPR p -> K=2p,2p+1 lanes 0-15; +16 lanes 16-31
                const int baseA = 2 * p + 16 * h;
                // B, same contiguous k order as A: VGPR p -> K=2p,2p+1 lanes 0-15; +16 lanes 16-31
                const int baseB = 2 * p + 16 * h;
#pragma unroll
                for (int e = 0; e < 2; ++e) {
                    const float pv = As[(baseA + e) * SS + mc];     // p in [0.01,0.99]
                    const float l1 = __logf(1.0f - pv);             // log(1-p)
                    const float av = __logf(pv) - l1;               // logit(p)
                    a[2 * p + e] = (_Float16)(valid ? av : 0.0f);
                    s1 += valid ? l1 : 0.0f;
                    const float bv = Bs[(baseB + e) * SS + mc];
                    bt[2 * p + e] = (_Float16)(valid ? bv : 0.0f);
                }
            }
            // EXEC is all-ones here (wave-uniform predicate), as WMMA requires.
            acc = __builtin_amdgcn_wmma_f32_16x16x32_f16(false, a, false, bt,
                                                         (short)0, acc, false, false);
        }
        __syncthreads();   // guard per-wave LDS slab reuse next iteration
    }

    // --- Deterministic cross-wave reduction through LDS (no float atomics) ---
#pragma unroll
    for (int r = 0; r < 8; ++r) sAcc[wave * 256 + r * 32 + lane] = acc[r];
    sS1[wave * 32 + lane] = valid ? s1 : 0.0f;
    __syncthreads();

    if (threadIdx.x < 256) {
        const int row = threadIdx.x >> 4;   // D row = r + 8*h
        const int n   = threadIdx.x & 15;
        const int hh  = row >> 3;
        const int r   = row & 7;
        float s = 0.0f;
#pragma unroll
        for (int w = 0; w < NWAVES; ++w) s += sAcc[w * 256 + r * 32 + n + 16 * hh];
        Mws[b * 256 + row * 16 + n] = s;
    } else if (threadIdx.x < 256 + 16) {
        const int mm = threadIdx.x - 256;
        float s = 0.0f;
#pragma unroll
        for (int w = 0; w < NWAVES; ++w)
            s += sS1[w * 32 + mm] + sS1[w * 32 + mm + 16];
        S1ws[b * 16 + mm] = s;
    }
}

// One thread per batch: 10x10 Jonker-Volgenant assignment + matched-cost sum.
__global__ __launch_bounds__(64)
void pit_hungarian_kernel(const float* __restrict__ Mws,
                          const float* __restrict__ S1ws,
                          float* __restrict__ out) {
    __shared__ float red[BB];
    const int b = threadIdx.x;

    float C[SS][SS];
    const float inv_t = 1.0f / (float)TT;
#pragma unroll
    for (int i = 0; i < SS; ++i) {
        const float s1 = S1ws[b * 16 + i];
#pragma unroll
        for (int j = 0; j < SS; ++j)
            C[i][j] = -(Mws[b * 256 + i * 16 + j] + s1) * inv_t;
    }

    float u[SS + 1], v[SS + 1];
    int   p[SS + 1], way[SS + 1];
    for (int i = 0; i <= SS; ++i) { u[i] = 0.0f; v[i] = 0.0f; p[i] = 0; way[i] = 0; }

    for (int i = 1; i <= SS; ++i) {
        p[0] = i;
        int j0 = 0;
        float minv[SS + 1];
        bool  used[SS + 1];
        for (int j = 0; j <= SS; ++j) { minv[j] = 1e30f; used[j] = false; }
        while (true) {
            used[j0] = true;
            const int i0 = p[j0];
            float delta = 1e30f;
            int j1 = 0;
            for (int j = 1; j <= SS; ++j) {
                if (!used[j]) {
                    const float cur = C[i0 - 1][j - 1] - u[i0] - v[j];
                    if (cur < minv[j]) { minv[j] = cur; way[j] = j0; }
                    if (minv[j] < delta) { delta = minv[j]; j1 = j; }
                }
            }
            for (int j = 0; j <= SS; ++j) {
                if (used[j]) { u[p[j]] += delta; v[j] -= delta; }
                else         { minv[j] -= delta; }
            }
            j0 = j1;
            if (p[j0] == 0) break;
        }
        while (j0) { const int j1 = way[j0]; p[j0] = p[j1]; j0 = j1; }
    }

    float s = 0.0f;
    for (int j = 1; j <= SS; ++j) s += C[p[j] - 1][j - 1];
    red[b] = s;
    __syncthreads();

    if (b == 0) {
        float tot = 0.0f;
        for (int k = 0; k < BB; ++k) tot += red[k];
        out[0] = tot / (float)(BB * SS);
    }
}

extern "C" void kernel_launch(void* const* d_in, const int* in_sizes, int n_in,
                              void* d_out, int out_size, void* d_ws, size_t ws_size,
                              hipStream_t stream) {
    const float* pred = (const float*)d_in[0];   // [64, 8000, 10] f32
    const float* tgt  = (const float*)d_in[1];   // [64, 8000, 10] f32

    float* Mws  = (float*)d_ws;                  // 64*256 floats
    float* S1ws = Mws + BB * 256;                // 64*16  floats (~68 KB of d_ws)

    pit_pairwise_wmma_kernel<<<BB, 32 * NWAVES, 0, stream>>>(pred, tgt, Mws, S1ws);
    pit_hungarian_kernel<<<1, BB, 0, stream>>>(Mws, S1ws, (float*)d_out);
}
